// DNN_1717986919052
// MI455X (gfx1250) — matrix operand order repaired and checked by reading, not compiled, not run on hardware
//
#include <hip/hip_runtime.h>
#include <cstdint>

typedef __attribute__((ext_vector_type(16))) _Float16 v16h;
typedef __attribute__((ext_vector_type(8)))  float    v8f;

#define H       256
#define DIN     39
#define ROWS_PB 128
#define NWAVE   8
#define WSTRIDE 264   // f16 elems per weight row in LDS (256 + 8 pad)
#define W1STRIDE 72   // f16 elems per padded-W1 row in LDS (64 + 8 pad)
#define ASTRIDE 264   // f16 elems per activation row in LDS
#define WBUF    (128 * WSTRIDE)

union AB { v16h v; uint4 q[2]; };

// Types matching the async-LDS builtin: AS1/AS3 pointers to a 16-byte int vector.
typedef int v4i_vec __attribute__((vector_size(16)));
typedef __attribute__((address_space(1))) v4i_vec* gp128_t;
typedef __attribute__((address_space(3))) v4i_vec* lp128_t;

__device__ __forceinline__ void copy16_g2l(const _Float16* g, _Float16* l) {
#if __has_builtin(__builtin_amdgcn_global_load_async_to_lds_b128)
  __builtin_amdgcn_global_load_async_to_lds_b128(
      (gp128_t)(uintptr_t)g, (lp128_t)(uint32_t)(uintptr_t)l, 0, 0);
#else
  *(uint4*)l = *(const uint4*)g;
#endif
}

template <int NCNT>
__device__ __forceinline__ void wait_async_lds() {
#if __has_builtin(__builtin_amdgcn_global_load_async_to_lds_b128)
#if __has_builtin(__builtin_amdgcn_s_wait_asynccnt)
  __builtin_amdgcn_s_wait_asynccnt(NCNT);
#else
  asm volatile("s_wait_asynccnt %0" :: "i"(NCNT) : "memory");
#endif
#endif
}

__device__ __forceinline__ float fast_tanh(float x) {
  float xc = fminf(fmaxf(x, -15.f), 15.f);
  float e = __expf(2.f * xc);
  return (e - 1.f) / (e + 1.f);
}

__device__ __forceinline__ float fast_elu(float v) {
  return v > 0.f ? v : (__expf(v) - 1.f);
}

// Row stats for LayerNorm over the in-register 16x256 tile.
// C layout: VGPR r holds row r (lanes 0-15) / row r+8 (lanes 16-31),
// col = 16*n + (lane&15). Reduce across the 16 tiles + 16 lanes of each half.
__device__ __forceinline__ void ln_stats(const v8f* C, float* mu, float* rs) {
  float s0[8], s1[8];
#pragma unroll
  for (int r = 0; r < 8; ++r) { s0[r] = 0.f; s1[r] = 0.f; }
#pragma unroll
  for (int n = 0; n < 16; ++n) {
#pragma unroll
    for (int r = 0; r < 8; ++r) { float v = C[n][r]; s0[r] += v; s1[r] += v * v; }
  }
#pragma unroll
  for (int m = 1; m < 16; m <<= 1) {
#pragma unroll
    for (int r = 0; r < 8; ++r) {
      s0[r] += __shfl_xor(s0[r], m, 32);
      s1[r] += __shfl_xor(s1[r], m, 32);
    }
  }
#pragma unroll
  for (int r = 0; r < 8; ++r) {
    float m_ = s0[r] * (1.f / 256.f);
    float var = s1[r] * (1.f / 256.f) - m_ * m_;
    mu[r] = m_;
    rs[r] = rsqrtf(var + 1e-5f);
  }
}

// Convert weights to f16 into workspace: [0,16384) = W1 padded 256x64, then W2..W7.
__global__ void prep_weights(const float* __restrict__ W1, const float* __restrict__ W2,
                             const float* __restrict__ W3, const float* __restrict__ W4,
                             const float* __restrict__ W5, const float* __restrict__ W6,
                             const float* __restrict__ W7, _Float16* __restrict__ wp) {
  int idx = blockIdx.x * 256 + threadIdx.x;
  if (idx < 16384) {
    int r = idx >> 6, c = idx & 63;
    float v = (c < DIN) ? W1[r * DIN + c] : 0.f;
    wp[idx] = (_Float16)v;
  } else if (idx < 409600) {
    int t = idx - 16384;
    int l = t >> 16, e = t & 65535;
    const float* W;
    switch (l) {
      case 0: W = W2; break; case 1: W = W3; break; case 2: W = W4; break;
      case 3: W = W5; break; case 4: W = W6; break; default: W = W7; break;
    }
    wp[idx] = (_Float16)W[e];
  }
}

__global__ __launch_bounds__(256, 1) void mlp_fused(
    const float* __restrict__ x, const _Float16* __restrict__ wp,
    const float* __restrict__ b1, const float* __restrict__ b2,
    const float* __restrict__ b3, const float* __restrict__ b4,
    const float* __restrict__ b5, const float* __restrict__ b6,
    const float* __restrict__ b7, const float* __restrict__ gamma,
    const float* __restrict__ beta, const float* __restrict__ W8,
    const float* __restrict__ b8, float* __restrict__ out, int N) {
  __shared__ __attribute__((aligned(16))) _Float16 wldsD[2 * WBUF];  // double buffer
  __shared__ __attribute__((aligned(16))) _Float16 actS[NWAVE][16 * ASTRIDE];
  __shared__ float gammaS[H], betaS[H], w8S[H];
  __shared__ float biasS[7][H];

  const int tid  = threadIdx.x;
  const int wave = tid >> 5, lane = tid & 31;
  const int half_ = lane >> 4, ln16 = lane & 15;
  const int rowBase = blockIdx.x * ROWS_PB + wave * 16;

  gammaS[tid] = gamma[tid]; betaS[tid] = beta[tid]; w8S[tid] = W8[tid];
  biasS[0][tid] = b1[tid]; biasS[1][tid] = b2[tid]; biasS[2][tid] = b3[tid];
  biasS[3][tid] = b4[tid]; biasS[4][tid] = b5[tid]; biasS[5][tid] = b6[tid];
  biasS[6][tid] = b7[tid];

  { // stage padded W1 (256x64 f16) into buffer 1, stride W1STRIDE (sync path)
    const uint4* src = (const uint4*)(wp + tid * 64);
    uint4* dst = (uint4*)(wldsD + WBUF + tid * W1STRIDE);
#pragma unroll
    for (int j = 0; j < 8; ++j) dst[j] = src[j];
  }

  _Float16* act = actS[wave];
  { // h0 = 0.1*log(x+1e-7), zero-padded 16x64 tile per wave
    _Float16* arow = act + ln16 * ASTRIDE;
    int row = rowBase + ln16;
#pragma unroll
    for (int j = 0; j < 20; ++j) {
      int col = half_ * 20 + j;
      float v = 0.f;
      if (col < DIN && row < N) v = 0.1f * __logf(x[row * DIN + col] + 1e-7f);
      if (col < 40) arow[col] = (_Float16)((col < DIN) ? v : 0.f);
    }
#pragma unroll
    for (int j = 0; j < 12; ++j) arow[40 + half_ * 12 + j] = (_Float16)0.f;
  }

  // prologue: async-stage weights for (layer 0, chunk 0) into buffer 0;
  // completes while fc1 computes out of buffer 1.
  {
    const _Float16* wsrc = wp + 16384;
#pragma unroll
    for (int rr = 0; rr < 16; ++rr) {
      int r = wave * 16 + rr;
      copy16_g2l(wsrc + r * 256 + lane * 8, wldsD + r * WSTRIDE + lane * 8);
    }
  }
  __syncthreads();

  v8f C[16];
#pragma unroll
  for (int n = 0; n < 16; ++n)
#pragma unroll
    for (int r = 0; r < 8; ++r) C[n][r] = 0.f;

  // ---- fc1 (K=64, weights in buffer 1 @ W1STRIDE) + tanh ----
#pragma unroll
  for (int k = 0; k < 2; ++k) {
    AB a;
    const _Float16* ap = act + ln16 * ASTRIDE + k * 32 + half_ * 8;
    a.q[0] = *(const uint4*)(ap);
    a.q[1] = *(const uint4*)(ap + 16);
#pragma unroll
    for (int n = 0; n < 16; ++n) {
      AB b;
      const _Float16* bp = wldsD + WBUF + (n * 16 + ln16) * W1STRIDE + k * 32 + half_ * 8;
      b.q[0] = *(const uint4*)(bp);
      b.q[1] = *(const uint4*)(bp + 16);
      C[n] = __builtin_amdgcn_wmma_f32_16x16x32_f16(false, a.v, false, b.v,
                                                    (short)0, C[n], false, false);
    }
  }
#pragma unroll
  for (int n = 0; n < 16; ++n) {
    float bb = biasS[0][n * 16 + ln16];
#pragma unroll
    for (int r = 0; r < 8; ++r) C[n][r] = fast_tanh(C[n][r] + bb);
  }

  // ---- fc2..fc7: h = elu(LN(h) @ W.T + b), double-buffered weights ----
#pragma unroll 1
  for (int l = 0; l < 6; ++l) {
    float mu[8], rs[8];
    ln_stats(C, mu, rs);
#pragma unroll
    for (int n = 0; n < 16; ++n) {
      int col = n * 16 + ln16;
      float g = gammaS[col], be = betaS[col];
#pragma unroll
      for (int r = 0; r < 8; ++r) {
        float v = (C[n][r] - mu[r]) * rs[r] * g + be;
        act[(r + half_ * 8) * ASTRIDE + col] = (_Float16)v;
      }
    }
#pragma unroll
    for (int n = 0; n < 16; ++n)
#pragma unroll
      for (int r = 0; r < 8; ++r) C[n][r] = 0.f;

    const _Float16* wbase = wp + 16384 + l * 65536;
    if (l < 5) __builtin_prefetch(wbase + 65536 + tid * 256, 0, 0);

#pragma unroll
    for (int chunk = 0; chunk < 2; ++chunk) {
      // buffer parity == chunk (stage = 2l+chunk)
      __syncthreads();  // all waves done with the buffer we are about to refill
      if (chunk == 0) {
        // issue async copy of (l, 1) into buffer 1 while we compute chunk 0
        const _Float16* wsrc = wbase + 32768;
#pragma unroll
        for (int rr = 0; rr < 16; ++rr) {
          int r = wave * 16 + rr;
          copy16_g2l(wsrc + r * 256 + lane * 8, wldsD + WBUF + r * WSTRIDE + lane * 8);
        }
        wait_async_lds<16>();  // (l,0) copy complete (in-order), (l,1) may remain
      } else {
        if (l < 5) {
          // issue async copy of (l+1, 0) into buffer 0 while we compute chunk 1
          const _Float16* wsrc = wbase + 65536;
#pragma unroll
          for (int rr = 0; rr < 16; ++rr) {
            int r = wave * 16 + rr;
            copy16_g2l(wsrc + r * 256 + lane * 8, wldsD + r * WSTRIDE + lane * 8);
          }
          wait_async_lds<16>();
        } else {
          wait_async_lds<0>();  // last stage: drain everything
        }
      }
      __syncthreads();

      const _Float16* wcur = wldsD + chunk * WBUF;
#pragma unroll
      for (int k = 0; k < 8; ++k) {
        AB a;
        const _Float16* ap = act + ln16 * ASTRIDE + k * 32 + half_ * 8;
        a.q[0] = *(const uint4*)(ap);
        a.q[1] = *(const uint4*)(ap + 16);
#pragma unroll
        for (int nn = 0; nn < 8; ++nn) {
          const int n = chunk * 8 + nn;
          AB b;
          const _Float16* bp = wcur + (nn * 16 + ln16) * WSTRIDE + k * 32 + half_ * 8;
          b.q[0] = *(const uint4*)(bp);
          b.q[1] = *(const uint4*)(bp + 16);
          C[n] = __builtin_amdgcn_wmma_f32_16x16x32_f16(false, a.v, false, b.v,
                                                        (short)0, C[n], false, false);
        }
      }
    }
#pragma unroll
    for (int n = 0; n < 16; ++n) {
      float bb = biasS[l + 1][n * 16 + ln16];
#pragma unroll
      for (int r = 0; r < 8; ++r) C[n][r] = fast_elu(C[n][r] + bb);
    }
  }

  // ---- final: LN then fc8 (256 -> 1), all in registers ----
  float mu[8], rs[8];
  ln_stats(C, mu, rs);
  float acc[8];
#pragma unroll
  for (int r = 0; r < 8; ++r) acc[r] = 0.f;
#pragma unroll
  for (int n = 0; n < 16; ++n) {
    int col = n * 16 + ln16;
    float g = gammaS[col], be = betaS[col], w = w8S[col];
#pragma unroll
    for (int r = 0; r < 8; ++r)
      acc[r] += ((C[n][r] - mu[r]) * rs[r] * g + be) * w;
  }
#pragma unroll
  for (int m = 1; m < 16; m <<= 1)
#pragma unroll
    for (int r = 0; r < 8; ++r) acc[r] += __shfl_xor(acc[r], m, 32);

  if (ln16 == 0) {
    float bb = b8[0];
#pragma unroll
    for (int r = 0; r < 8; ++r) {
      int row = rowBase + half_ * 8 + r;
      if (row < N) out[row] = acc[r] + bb;
    }
  }
}

extern "C" void kernel_launch(void* const* d_in, const int* in_sizes, int n_in,
                              void* d_out, int out_size, void* d_ws, size_t ws_size,
                              hipStream_t stream) {
  const float* x     = (const float*)d_in[0];
  const float* W1    = (const float*)d_in[1];
  const float* b1    = (const float*)d_in[2];
  const float* W2    = (const float*)d_in[3];
  const float* b2    = (const float*)d_in[4];
  const float* W3    = (const float*)d_in[5];
  const float* b3    = (const float*)d_in[6];
  const float* W4    = (const float*)d_in[7];
  const float* b4    = (const float*)d_in[8];
  const float* W5    = (const float*)d_in[9];
  const float* b5    = (const float*)d_in[10];
  const float* W6    = (const float*)d_in[11];
  const float* b6    = (const float*)d_in[12];
  const float* W7    = (const float*)d_in[13];
  const float* b7    = (const float*)d_in[14];
  const float* W8    = (const float*)d_in[15];
  const float* b8    = (const float*)d_in[16];
  const float* gamma = (const float*)d_in[17];
  const float* beta  = (const float*)d_in[18];
  float* out = (float*)d_out;

  _Float16* wp = (_Float16*)d_ws;
  const int N = in_sizes[0] / DIN;

  prep_weights<<<(409600 + 255) / 256, 256, 0, stream>>>(W1, W2, W3, W4, W5, W6, W7, wp);
  mlp_fused<<<(N + ROWS_PB - 1) / ROWS_PB, 256, 0, stream>>>(
      x, wp, b1, b2, b3, b4, b5, b6, b7, gamma, beta, W8, b8, out, N);
}
